// Transformer_Block_57183194579226
// MI455X (gfx1250) — matrix operand order repaired and checked by reading, not compiled, not run on hardware
//
#include <hip/hip_runtime.h>
#include <hip/hip_bf16.h>
#include <math.h>

typedef __attribute__((ext_vector_type(16))) __bf16 v16bf;
typedef __attribute__((ext_vector_type(8)))  float  v8f;

#define NTOK   4096
#define NBATCH 32

__device__ __forceinline__ float fast_silu(float s) {
  // s * sigmoid(s) with v_rcp_f32 instead of the IEEE divide sequence
  return s * __builtin_amdgcn_rcpf(1.f + __expf(-s));
}

// ---------------------------------------------------------------------------
// GEMM (NT): out[m,n] = SILU ? silu( res + A.Wt + b ) : ( A.Wt + b )
//   A : [32, 4096] row-major (K contiguous)
//   W : [4096, 4096] row-major (K contiguous) -> B = W^T, column n = W row n
// Each wave computes a 32(M) x 16(N) tile via two v_wmma_f32_16x16x32_bf16.
// 4 waves / block, grid.x = 64  ->  covers N = 4096.
// ---------------------------------------------------------------------------
template <bool SILU>
__global__ __launch_bounds__(128) void tb_gemm_nt(
    const float* __restrict__ A, const float* __restrict__ W,
    const float* __restrict__ bias, const float* __restrict__ res,
    float* __restrict__ out)
{
  const int lane = threadIdx.x & 31;
  const int wave = threadIdx.x >> 5;
  const int r    = lane & 15;   // row within 16-tile (A) / column (B,D)
  const int half = lane >> 4;   // lane-half selects K sub-chunk
  const int n0   = (blockIdx.x * 4 + wave) * 16;

  const float* a0p = A + (size_t)r * NTOK;          // A rows 0..15
  const float* a1p = A + (size_t)(r + 16) * NTOK;   // A rows 16..31
  const float* wbp = W + (size_t)(n0 + r) * NTOK;   // W row = B column
  const int koffA = half * 8;    // A frag: K pairs interleaved at 8-granularity
  const int koffB = half * 8;    // B frag: K interleaved at 8-granularity (matches A)

  v8f acc0 = {0.f,0.f,0.f,0.f,0.f,0.f,0.f,0.f};
  v8f acc1 = {0.f,0.f,0.f,0.f,0.f,0.f,0.f,0.f};

  for (int k0 = 0; k0 < NTOK; k0 += 32) {
    __builtin_prefetch(wbp + k0 + 256, 0, 1);   // global_prefetch_b8
    v16bf fa0, fa1, fb;
#pragma unroll
    for (int i = 0; i < 8; ++i) {
      fa0[i]     = (__bf16)a0p[k0 + koffA + i];
      fa0[i + 8] = (__bf16)a0p[k0 + 16 + koffA + i];
      fa1[i]     = (__bf16)a1p[k0 + koffA + i];
      fa1[i + 8] = (__bf16)a1p[k0 + 16 + koffA + i];
    }
#pragma unroll
    for (int i = 0; i < 8; ++i) {
      fb[i]     = (__bf16)wbp[k0 + koffB + i];
      fb[i + 8] = (__bf16)wbp[k0 + 16 + koffB + i];
    }

    acc0 = __builtin_amdgcn_wmma_f32_16x16x32_bf16(false, fa0, false, fb,
                                                   (short)0, acc0, false, false);
    acc1 = __builtin_amdgcn_wmma_f32_16x16x32_bf16(false, fa1, false, fb,
                                                   (short)0, acc1, false, false);
  }

  // D layout: VGPR i -> M = i + 8*half ; column = n0 + (lane & 15)
  const int   n  = n0 + r;
  const float bn = bias[n];
#pragma unroll
  for (int i = 0; i < 8; ++i) {
    {
      const int m = i + 8 * half;
      float y = acc0[i] + bn;
      if (SILU) y = fast_silu(y + res[(size_t)m * NTOK + n]);
      out[(size_t)m * NTOK + n] = y;
    }
    {
      const int m = 16 + i + 8 * half;
      float y = acc1[i] + bn;
      if (SILU) y = fast_silu(y + res[(size_t)m * NTOK + n]);
      out[(size_t)m * NTOK + n] = y;
    }
  }
}

// ---------------------------------------------------------------------------
// Rank-1 attention + silu(x + attended).
// One block = (batch b, 256 consecutive i). k[b,:], v[b,:] staged into LDS
// with GLOBAL_LOAD_ASYNC_TO_LDS_B128 (ASYNCcnt path); per-batch k max/min
// reduced from the staged copy for an exact softmax max.
// ---------------------------------------------------------------------------
__global__ __launch_bounds__(256) void tb_attn(
    const float* __restrict__ q, const float* __restrict__ k,
    const float* __restrict__ v, const float* __restrict__ x,
    float* __restrict__ h)
{
  __shared__ float kl[NTOK];
  __shared__ float vl[NTOK];
  __shared__ float red[512];

  const int b   = blockIdx.y;
  const int tid = threadIdx.x;
  const float* kb = k + (size_t)b * NTOK;
  const float* vb = v + (size_t)b * NTOK;

  // ---- async stage k,v -> LDS (16 floats per thread per array) ----
  for (int j = tid * 4; j < NTOK; j += 256 * 4) {
    unsigned lk = (unsigned)(unsigned long long)(kl + j);  // LDS byte offset
    unsigned lv = (unsigned)(unsigned long long)(vl + j);
    unsigned long long gk = (unsigned long long)(kb + j);
    unsigned long long gv = (unsigned long long)(vb + j);
    asm volatile("global_load_async_to_lds_b128 %0, %1, off"
                 :: "v"(lk), "v"(gk) : "memory");
    asm volatile("global_load_async_to_lds_b128 %0, %1, off"
                 :: "v"(lv), "v"(gv) : "memory");
  }
  asm volatile("s_wait_asynccnt 0x0" ::: "memory");
  __syncthreads();

  // ---- per-batch k max/min from the staged copy ----
  float mx = -1e30f, mn = 1e30f;
  for (int j = tid * 4; j < NTOK; j += 256 * 4) {
    float4 kk = *(const float4*)(kl + j);
    mx = fmaxf(mx, fmaxf(fmaxf(kk.x, kk.y), fmaxf(kk.z, kk.w)));
    mn = fminf(mn, fminf(fminf(kk.x, kk.y), fminf(kk.z, kk.w)));
  }
  red[tid] = mx; red[256 + tid] = mn;
  __syncthreads();
  for (int s = 128; s > 0; s >>= 1) {
    if (tid < s) {
      red[tid]       = fmaxf(red[tid], red[tid + s]);
      red[256 + tid] = fminf(red[256 + tid], red[256 + tid + s]);
    }
    __syncthreads();
  }
  const float kmax = red[0], kmin = red[256];

  const int   i = blockIdx.x * 256 + tid;
  const float t = q[(size_t)b * NTOK + i];
  const float m = fmaxf(t * kmax, t * kmin);   // exact rowwise max of t*k_j

  float den = 0.f, num = 0.f;
  for (int j = 0; j < NTOK; j += 4) {
    const float4 kk = *(const float4*)(kl + j);   // ds_load_b128, broadcast
    const float4 vv = *(const float4*)(vl + j);
    float e;
    e = __expf(t * kk.x - m); den += e; num = fmaf(e, vv.x, num);
    e = __expf(t * kk.y - m); den += e; num = fmaf(e, vv.y, num);
    e = __expf(t * kk.z - m); den += e; num = fmaf(e, vv.z, num);
    e = __expf(t * kk.w - m); den += e; num = fmaf(e, vv.w, num);
  }
  const float att = num * __builtin_amdgcn_rcpf(den);
  h[(size_t)b * NTOK + i] = fast_silu(x[(size_t)b * NTOK + i] + att);
}

// ---------------------------------------------------------------------------
extern "C" void kernel_launch(void* const* d_in, const int* in_sizes, int n_in,
                              void* d_out, int out_size, void* d_ws, size_t ws_size,
                              hipStream_t stream)
{
  const float* x  = (const float*)d_in[0];
  const float* Wq = (const float*)d_in[1];
  const float* bq = (const float*)d_in[2];
  const float* Wk = (const float*)d_in[3];
  const float* bk = (const float*)d_in[4];
  const float* Wv = (const float*)d_in[5];
  const float* bv = (const float*)d_in[6];
  const float* Wp = (const float*)d_in[7];
  const float* bp = (const float*)d_in[8];
  float* out = (float*)d_out;

  float* qw = (float*)d_ws;                 // [32,4096]
  float* kw = qw + (size_t)NBATCH * NTOK;   // [32,4096]
  float* vw = kw + (size_t)NBATCH * NTOK;   // [32,4096]
  float* hw = vw + (size_t)NBATCH * NTOK;   // [32,4096]

  const dim3 gg(64), gb(128);
  tb_gemm_nt<false><<<gg, gb, 0, stream>>>(x, Wq, bq, nullptr, qw);
  tb_gemm_nt<false><<<gg, gb, 0, stream>>>(x, Wk, bk, nullptr, kw);
  tb_gemm_nt<false><<<gg, gb, 0, stream>>>(x, Wv, bv, nullptr, vw);

  tb_attn<<<dim3(NTOK / 256, NBATCH), 256, 0, stream>>>(qw, kw, vw, x, hw);

  tb_gemm_nt<true><<<gg, gb, 0, stream>>>(hw, Wp, bp, hw, out);
}
